// Dynamic_deformable_DySample_restart_40338332844084
// MI455X (gfx1250) — matrix operand order repaired and checked by reading, not compiled, not run on hardware
//
#include <hip/hip_runtime.h>

typedef __attribute__((ext_vector_type(16))) _Float16 v16h;
typedef __attribute__((ext_vector_type(8)))  _Float16 v8h;
typedef __attribute__((ext_vector_type(8)))  float    v8f;

#define Hc   352
#define Wc_  1216
#define Bc   4
#define CIN  30
#define COUT 84
#define NPAD 96
#define TILEX 256
#define GW   258     // halo width: TILEX + 2
#define NSLAB 9      // (r,s) slabs; K-tile = one slab of 32 (30 real c + 2 pad)
#define CS   32      // padded channels per slab

static const long P_TOT = (long)Bc * Hc * Wc_;   // 1,712,128

// ---------------------------------------------------------------------------
// Kernel 1: fused 3x3 conv (implicit GEMM via v_wmma_f32_16x16x32_f16)
//           + bias + per-step softmax + fp16 quantize.
// Block: 256 threads (8 waves). Each block: 256 pixels of one row, all 84 ch.
// Each wave owns two 16-row M-tiles so every B fragment feeds 2 WMMAs.
// ---------------------------------------------------------------------------
__global__ __launch_bounds__(256) void conv_offsets_kernel(
    const float* __restrict__ guidance,   // (B,30,H,W)
    const float* __restrict__ Wconv,      // (84,30,3,3)
    const float* __restrict__ bconv,      // (84)
    _Float16*    __restrict__ offbuf,     // (84, P) channel-major fp16
    long P)
{
  extern __shared__ char smem[];
  _Float16* Ws = (_Float16*)smem;               // [NSLAB][NPAD][CS] = 27648 h
  _Float16* Gs = Ws + NSLAB * NPAD * CS;        // [3][GW][CS]      = 24768 h
  float*    Cs = (float*)smem;                  // [256][96] aliases staging

  const int tid = threadIdx.x;
  const int b   = blockIdx.z;
  const int y   = blockIdx.y;
  const int x0  = blockIdx.x * TILEX;

  // ---- stage weights: W'[slab][n][c] = Wconv[n][c][r][s], fp16, zero-padded
  for (int i = tid; i < NSLAB * NPAD * CS; i += 256) {
    int c    = i & (CS - 1);
    int n    = (i >> 5) % NPAD;
    int slab = i / (NPAD * CS);
    float v = 0.0f;
    if (c < CIN && n < COUT) {
      int r = slab / 3, s = slab % 3;
      v = Wconv[((n * CIN + c) * 3 + r) * 3 + s];
    }
    Ws[i] = (_Float16)v;
  }
  // ---- stage guidance halo: G[r][xl][c] = guidance[b][c][y-1+r][x0-1+xl]
  for (int i = tid; i < 3 * GW * CS; i += 256) {
    int c  = i & (CS - 1);
    int xl = (i >> 5) % GW;
    int r  = i / (GW * CS);
    int yy = y - 1 + r;
    int xx = x0 - 1 + xl;
    float v = 0.0f;
    if (c < CIN && yy >= 0 && yy < Hc && xx >= 0 && xx < Wc_)
      v = guidance[(((long)b * CIN + c) * Hc + yy) * Wc_ + xx];
    Gs[i] = (_Float16)v;
  }
  __syncthreads();

  // ---- implicit GEMM: C[256 x 96] += A[256 x 288] * B[288 x 96]
  const int lane    = tid & 31;
  const int wave    = tid >> 5;
  const int half_id = lane >> 4;   // 0: lanes 0-15, 1: lanes 16-31
  const int l       = lane & 15;
  const int m0      = wave * 32;   // this wave's 32-pixel strip

  union frag { v16h v; v8h h[2]; };

  v8f acc0[6], acc1[6];
#pragma unroll
  for (int nt = 0; nt < 6; ++nt) {
    acc0[nt] = (v8f){0,0,0,0,0,0,0,0};
    acc1[nt] = (v8f){0,0,0,0,0,0,0,0};
  }

#pragma unroll
  for (int slab = 0; slab < NSLAB; ++slab) {
    const int r = slab / 3, s = slab % 3;
    // A fragments 16x32: VGPR0-3 = K[half*8..+7], VGPR4-7 = K[16+half*8..+7]
    const _Float16* gp0 = &Gs[((r * GW) + (m0 + l + s)) * CS];
    const _Float16* gp1 = gp0 + 16 * CS;
    frag a0, a1, bf[6];
    a0.h[0] = *(const v8h*)(gp0 + half_id * 8);
    a0.h[1] = *(const v8h*)(gp0 + 16 + half_id * 8);
    a1.h[0] = *(const v8h*)(gp1 + half_id * 8);
    a1.h[1] = *(const v8h*)(gp1 + 16 + half_id * 8);
    // B fragments 32x16: VGPR0-3 = K[half*8..+7], VGPR4-7 = K[16+half*8..+7]
#pragma unroll
    for (int nt = 0; nt < 6; ++nt) {
      const _Float16* wp = &Ws[((slab * NPAD) + nt * 16 + l) * CS + half_id * 8];
      bf[nt].h[0] = *(const v8h*)(wp);
      bf[nt].h[1] = *(const v8h*)(wp + 16);
    }
#pragma unroll
    for (int nt = 0; nt < 6; ++nt) {
      acc0[nt] = __builtin_amdgcn_wmma_f32_16x16x32_f16(
          false, a0.v, false, bf[nt].v, (short)0, acc0[nt], false, false);
      acc1[nt] = __builtin_amdgcn_wmma_f32_16x16x32_f16(
          false, a1.v, false, bf[nt].v, (short)0, acc1[nt], false, false);
    }
  }

  // ---- dump C to LDS (aliases staging; wait until all waves' reads done)
  __syncthreads();
#pragma unroll
  for (int nt = 0; nt < 6; ++nt) {
    int n = nt * 16 + l;
#pragma unroll
    for (int v = 0; v < 8; ++v) {
      int row0 = m0 + half_id * 8 + v;        // C layout: M = v (+8 hi lanes)
      Cs[row0 * NPAD + n]        = acc0[nt][v];
      Cs[(row0 + 16) * NPAD + n] = acc1[nt][v];
    }
  }
  __syncthreads();

  // ---- epilogue: bias + 3x (18 offsets, softmax-10 -> 9 aff + restart), fp16
  {
    int x = x0 + tid;
    if (x < Wc_) {
      long pix = ((long)b * Hc + y) * Wc_ + x;
#pragma unroll
      for (int t = 0; t < 3; ++t) {
        float v[28];
#pragma unroll
        for (int cc = 0; cc < 28; ++cc)
          v[cc] = Cs[tid * NPAD + t * 28 + cc] + bconv[t * 28 + cc];
        // raw offsets (dy/dx interleaved), fp16-rounded like the reference
#pragma unroll
        for (int j = 0; j < 18; ++j)
          offbuf[(long)(t * 28 + j) * P + pix] = (_Float16)v[j];
        // softmax over v[18..27]
        float mx = v[18];
#pragma unroll
        for (int j = 1; j < 10; ++j) mx = fmaxf(mx, v[18 + j]);
        float e[10], sum = 0.0f;
#pragma unroll
        for (int j = 0; j < 10; ++j) { e[j] = __expf(v[18 + j] - mx); sum += e[j]; }
        float inv = 1.0f / sum;
#pragma unroll
        for (int j = 0; j < 10; ++j)
          offbuf[(long)(t * 28 + 18 + j) * P + pix] = (_Float16)(e[j] * inv);
      }
    }
  }
}

// ---------------------------------------------------------------------------
// Kernel 2: one deformable-propagation step (scalar gather kernel).
// ---------------------------------------------------------------------------
__device__ __forceinline__ float dcv_sample(const float* __restrict__ f2d,
                                            int yi, int xi) {
  if (yi >= 0 && yi < Hc && xi >= 0 && xi < Wc_)
    return f2d[(long)yi * Wc_ + xi];
  return 0.0f;
}

__global__ __launch_bounds__(256) void prop_step_kernel(
    const float* __restrict__ feat_in,
    const float* __restrict__ feat_fix,
    const float* __restrict__ confidence,
    const _Float16* __restrict__ offbuf,
    float* __restrict__ feat_out,
    int step, long P)
{
  long idx = (long)blockIdx.x * 256 + threadIdx.x;
  if (idx >= P) return;
  int x = (int)(idx % Wc_);
  long rem = idx / Wc_;
  int y = (int)(rem % Hc);
  int b = (int)(rem / Hc);

  const float* f2d = feat_in + (long)b * Hc * Wc_;
  const _Float16* ob = offbuf + (long)(step * 28) * P;

  float ff = feat_fix[idx];
  float sg = (ff > 0.0f ? 1.0f : 0.0f) - (ff < 0.0f ? 1.0f : 0.0f);
  float conf = sg * (1.0f / (1.0f + __expf(-confidence[idx])));

  float prop = 0.0f;
#pragma unroll
  for (int tap = 0; tap < 9; ++tap) {
    float dy  = (float)ob[(long)(2 * tap)     * P + idx];
    float dx  = (float)ob[(long)(2 * tap + 1) * P + idx];
    float aff = (float)ob[(long)(18 + tap)    * P + idx];
    float ys = (float)(y - 1 + tap / 3) + dy;
    float xs = (float)(x - 1 + tap % 3) + dx;
    float y0f = floorf(ys), x0f = floorf(xs);
    int   y0 = (int)y0f,    xq = (int)x0f;
    float wy = ys - y0f,    wx = xs - x0f;
    float val = dcv_sample(f2d, y0,     xq    ) * (1.0f - wy) * (1.0f - wx)
              + dcv_sample(f2d, y0,     xq + 1) * (1.0f - wy) * wx
              + dcv_sample(f2d, y0 + 1, xq    ) * wy * (1.0f - wx)
              + dcv_sample(f2d, y0 + 1, xq + 1) * wy * wx;
    prop += val * aff;
  }
  float restart = (float)ob[(long)27 * P + idx];
  float fi = feat_in[idx];
  feat_out[idx] = (1.0f - conf) * (prop + restart * fi) + conf * ff;
}

// ---------------------------------------------------------------------------
extern "C" void kernel_launch(void* const* d_in, const int* in_sizes, int n_in,
                              void* d_out, int out_size, void* d_ws, size_t ws_size,
                              hipStream_t stream) {
  const float* feat_init  = (const float*)d_in[0];
  const float* guidance   = (const float*)d_in[1];
  const float* confidence = (const float*)d_in[2];
  const float* feat_fix   = (const float*)d_in[3];
  const float* Wconv      = (const float*)d_in[4];
  const float* bconv      = (const float*)d_in[5];
  float* out = (float*)d_out;

  const long P = P_TOT;
  _Float16* offbuf = (_Float16*)d_ws;                       // 84*P halves
  size_t offbytes = ((size_t)84 * P * sizeof(_Float16) + 255) & ~(size_t)255;
  float* featA = (float*)((char*)d_ws + offbytes);
  float* featB = featA + P;

  // LDS: weights (27648h) + guidance halo (24768h) = 104832 B (C-dump aliases)
  const size_t lds_bytes =
      (size_t)(NSLAB * NPAD * CS + 3 * GW * CS) * sizeof(_Float16);

  dim3 g1((Wc_ + TILEX - 1) / TILEX, Hc, Bc);
  conv_offsets_kernel<<<g1, 256, lds_bytes, stream>>>(guidance, Wconv, bconv,
                                                      offbuf, P);

  dim3 g2((unsigned)((P + 255) / 256));
  prop_step_kernel<<<g2, 256, 0, stream>>>(feat_init, feat_fix, confidence,
                                           offbuf, featA, 0, P);
  prop_step_kernel<<<g2, 256, 0, stream>>>(featA, feat_fix, confidence,
                                           offbuf, featB, 1, P);
  prop_step_kernel<<<g2, 256, 0, stream>>>(featB, feat_fix, confidence,
                                           offbuf, out, 2, P);
}
